// TopKSAE_29008209117481
// MI455X (gfx1250) — compile-verified, matrix operand order checked by reading, not run on hardware
//
#include <hip/hip_runtime.h>
#include <hip/hip_bf16.h>
#include <stdint.h>

#define N_ROWS  4096
#define DM      2048
#define DS      32768
#define K_TOP   64

typedef __bf16 v16bf __attribute__((ext_vector_type(16)));
typedef float  v8f   __attribute__((ext_vector_type(8)));

// CDNA5 async global->LDS path (ASYNCcnt) if the toolchain declares it.
#if defined(__has_builtin)
#if __has_builtin(__builtin_amdgcn_global_load_async_to_lds_b128) && \
    __has_builtin(__builtin_amdgcn_s_wait_asynccnt)
#define USE_ASYNC_LDS 1
#endif
#endif
#ifndef USE_ASYNC_LDS
#define USE_ASYNC_LDS 0
#endif

#if USE_ASYNC_LDS
// Exact parameter types of the async-copy builtin (from clang's declaration):
// (v4i addrspace(1)*, v4i addrspace(3)*, imm offset, imm cpol)
typedef int v4i_vs __attribute__((vector_size(16)));
typedef __attribute__((address_space(1))) v4i_vs* g_b128_ptr;
typedef __attribute__((address_space(3))) v4i_vs* l_b128_ptr;
#endif

// ---------------------------------------------------------------- helpers
static __device__ __forceinline__ uint32_t pack_bf16x2(float lo, float hi) {
    union { float f; uint32_t u; } a{lo}, b{hi};
    uint32_t ua = a.u + 0x7FFFu + ((a.u >> 16) & 1u);   // RNE
    uint32_t ub = b.u + 0x7FFFu + ((b.u >> 16) & 1u);
    return (ua >> 16) | (ub & 0xFFFF0000u);
}

static __device__ __forceinline__ unsigned fkey(float f) {
    union { float f; unsigned u; } c{f};
    unsigned u = c.u;
    return u ^ ((u & 0x80000000u) ? 0xFFFFFFFFu : 0x80000000u);
}

// ------------------------------------------------- 1) f32 -> bf16 (+bias sub)
__global__ __launch_bounds__(256)
void convert_bf16_kernel(const float* __restrict__ src,
                         const float* __restrict__ bias,   // per-column, may be null
                         uint32_t* __restrict__ dst,       // packed 2x bf16
                         int cols, long long total) {
    long long i = ((long long)blockIdx.x * blockDim.x + threadIdx.x) * 8;
    if (i >= total) return;
    float4 a = *(const float4*)(src + i);
    float4 b = *(const float4*)(src + i + 4);
    if (bias) {
        int col = (int)(i & (cols - 1));                  // cols is power of two
        float4 p = *(const float4*)(bias + col);
        float4 q = *(const float4*)(bias + col + 4);
        a.x -= p.x; a.y -= p.y; a.z -= p.z; a.w -= p.w;
        b.x -= q.x; b.y -= q.y; b.z -= q.z; b.w -= q.w;
    }
    uint4 o;
    o.x = pack_bf16x2(a.x, a.y);
    o.y = pack_bf16x2(a.z, a.w);
    o.z = pack_bf16x2(b.x, b.y);
    o.w = pack_bf16x2(b.z, b.w);
    *(uint4*)(dst + (i >> 1)) = o;
}

// ------------------------------------------------- 2) encode GEMM (bf16 WMMA)
// z_pre[N][DS] = A[N][DM](bf16) * W[DS][DM](bf16)^T + b_enc
// Block: 256 thr = 8 waves, arranged 2(M) x 4(S); wave tile = 32M x 32S
// (4 accumulators -> 2 loads/WMMA, 4 independent WMMA chains).
__global__ __launch_bounds__(256)
void encode_wmma_kernel(const uint16_t* __restrict__ A,
                        const uint16_t* __restrict__ W,
                        const float* __restrict__ b_enc,
                        float* __restrict__ z_pre) {
    const int wave  = threadIdx.x >> 5;
    const int lane  = threadIdx.x & 31;
    const int waveM = wave >> 2;          // 0..1
    const int waveS = wave & 3;           // 0..3
    const int m0 = blockIdx.y * 64 + waveM * 32;
    const int s0 = blockIdx.x * 128 + waveS * 32;
    const int r    = lane & 15;
    const int koff = (lane >> 4) << 3;    // 0 or 8 (16-bit A/B VGPR layout)

    const uint16_t* aRow0 = A + (size_t)(m0 + r) * DM + koff;
    const uint16_t* aRow1 = aRow0 + (size_t)16 * DM;
    const uint16_t* wRow0 = W + (size_t)(s0 + r) * DM + koff;
    const uint16_t* wRow1 = wRow0 + (size_t)16 * DM;

    v8f acc00 = {}, acc01 = {}, acc10 = {}, acc11 = {};
    union Frag { uint4 q[2]; v16bf v; };

    for (int k0 = 0; k0 < DM; k0 += 32) {
        __builtin_prefetch(aRow0 + k0 + 512, 0, 1);   // global_prefetch_b8
        __builtin_prefetch(aRow1 + k0 + 512, 0, 1);
        __builtin_prefetch(wRow0 + k0 + 512, 0, 1);
        __builtin_prefetch(wRow1 + k0 + 512, 0, 1);
        Frag a0, a1, b0, b1;
        a0.q[0] = *(const uint4*)(aRow0 + k0);
        a0.q[1] = *(const uint4*)(aRow0 + k0 + 16);
        a1.q[0] = *(const uint4*)(aRow1 + k0);
        a1.q[1] = *(const uint4*)(aRow1 + k0 + 16);
        b0.q[0] = *(const uint4*)(wRow0 + k0);
        b0.q[1] = *(const uint4*)(wRow0 + k0 + 16);
        b1.q[0] = *(const uint4*)(wRow1 + k0);
        b1.q[1] = *(const uint4*)(wRow1 + k0 + 16);
        acc00 = __builtin_amdgcn_wmma_f32_16x16x32_bf16(false, a0.v, false, b0.v,
                                                        (short)0, acc00, false, false);
        acc01 = __builtin_amdgcn_wmma_f32_16x16x32_bf16(false, a0.v, false, b1.v,
                                                        (short)0, acc01, false, false);
        acc10 = __builtin_amdgcn_wmma_f32_16x16x32_bf16(false, a1.v, false, b0.v,
                                                        (short)0, acc10, false, false);
        acc11 = __builtin_amdgcn_wmma_f32_16x16x32_bf16(false, a1.v, false, b1.v,
                                                        (short)0, acc11, false, false);
    }

    // C/D layout: VGPR j -> M = j + 8*(lane>=16), N = lane&15
    const int hi    = lane >> 4;
    const int scol  = s0 + (lane & 15);
    const float be0 = b_enc[scol];
    const float be1 = b_enc[scol + 16];
    float* zp0 = z_pre + (size_t)(m0 + 8 * hi) * DS;
    float* zp1 = zp0 + (size_t)16 * DS;
#pragma unroll
    for (int j = 0; j < 8; ++j) {
        zp0[(size_t)j * DS + scol]      = acc00[j] + be0;
        zp0[(size_t)j * DS + scol + 16] = acc01[j] + be1;
        zp1[(size_t)j * DS + scol]      = acc10[j] + be0;
        zp1[(size_t)j * DS + scol + 16] = acc11[j] + be1;
    }
}

// ------------------------------------------------- 3) top-K radix select
// One 256-thread block per row. The 128KB row is staged once into LDS
// (CDNA5 async global->LDS when available), then 4x radix-256 MSB-first
// select finds the K-th largest key; a final pass writes the full z_sparse
// row and compacts the K (value,index) pairs for the sparse decode.
__global__ __launch_bounds__(256)
void topk_kernel(const float* __restrict__ z_pre,
                 float* __restrict__ z_sparse,
                 float* __restrict__ tv,
                 int* __restrict__ ti) {
    extern __shared__ float rowv[];                   // DS floats (128 KB)
    __shared__ unsigned hist[256];
    __shared__ unsigned sh_prefix, sh_want, sh_eq, sh_slot;

    const int n   = blockIdx.x;
    const int tid = threadIdx.x;
    const float* zr = z_pre    + (size_t)n * DS;
    float*       zs = z_sparse + (size_t)n * DS;

    // Stage row into LDS (one pass over HBM instead of five).
    for (int i = tid * 4; i < DS; i += 256 * 4) {
#if USE_ASYNC_LDS
        __builtin_amdgcn_global_load_async_to_lds_b128(
            (g_b128_ptr)(zr + i), (l_b128_ptr)(rowv + i), 0, 0);
#else
        *(float4*)(rowv + i) = *(const float4*)(zr + i);
#endif
    }
#if USE_ASYNC_LDS
    __builtin_amdgcn_s_wait_asynccnt(0);
#endif
    __syncthreads();

    unsigned prefix = 0, want = K_TOP, maskAbove = 0;
#pragma unroll
    for (int round = 0; round < 4; ++round) {
        const int shift = 24 - 8 * round;
        hist[tid] = 0;
        __syncthreads();
        for (int i = tid; i < DS; i += 256) {
            unsigned u = fkey(rowv[i]);
            if ((u & maskAbove) == prefix)
                atomicAdd(&hist[(u >> shift) & 255u], 1u);
        }
        __syncthreads();
        if (tid == 0) {
            unsigned cum = 0; int b = 255;
            for (; b > 0; --b) { cum += hist[b]; if (cum >= want) break; }
            sh_want   = want - (cum - hist[b]);
            sh_prefix = prefix | ((unsigned)b << shift);
        }
        __syncthreads();
        prefix = sh_prefix; want = sh_want;
        maskAbove = 0xFFFFFFFFu << shift;
        __syncthreads();
    }
    const unsigned uth = prefix;      // K-th largest key
    const unsigned weq = want;        // how many ==uth to keep

    if (tid == 0) { sh_eq = 0; sh_slot = 0; }
    __syncthreads();

    for (int i = tid; i < DS; i += 256) {
        float v = rowv[i];
        unsigned u = fkey(v);
        bool sel = (u > uth);
        if (!sel && u == uth) {
            unsigned p = atomicAdd(&sh_eq, 1u);
            sel = (p < weq);
        }
        zs[i] = sel ? v : 0.0f;
        if (sel) {
            unsigned s = atomicAdd(&sh_slot, 1u);
            tv[(size_t)n * K_TOP + s] = v;
            ti[(size_t)n * K_TOP + s] = i;
        }
    }
}

// ------------------------------------------------- 4) W_dec transpose
// [DM][DS] -> [DS][DM] so decode gathers are coalesced rows.
__global__ __launch_bounds__(256)
void transpose_kernel(const float* __restrict__ src, float* __restrict__ dst) {
    __shared__ float tile[32][33];
    const int x  = blockIdx.x * 32 + threadIdx.x;   // col in src (s)
    const int y0 = blockIdx.y * 32;                 // row in src (d)
#pragma unroll
    for (int j = threadIdx.y; j < 32; j += 8)
        tile[j][threadIdx.x] = src[(size_t)(y0 + j) * DS + x];
    __syncthreads();
    const int xo = blockIdx.y * 32 + threadIdx.x;   // col in dst (d)
    const int yo = blockIdx.x * 32;                 // row in dst (s)
#pragma unroll
    for (int j = threadIdx.y; j < 32; j += 8)
        dst[(size_t)(yo + j) * DM + xo] = tile[threadIdx.x][j];
}

// ------------------------------------------------- 5) sparse decode
// x_hat[n][:] = sum_k tv[n,k] * W_decT[ti[n,k]][:] + b_dec
__global__ __launch_bounds__(256)
void decode_kernel(const float* __restrict__ wdT,
                   const float* __restrict__ tv,
                   const int* __restrict__ ti,
                   const float* __restrict__ b_dec,
                   float* __restrict__ x_hat) {
    const int n  = blockIdx.x;
    const int d0 = threadIdx.x * 8;                 // 256*8 == DM
    float4 acc0 = *(const float4*)(b_dec + d0);
    float4 acc1 = *(const float4*)(b_dec + d0 + 4);
#pragma unroll 4
    for (int k = 0; k < K_TOP; ++k) {
        const float v = tv[(size_t)n * K_TOP + k];
        const int   c = ti[(size_t)n * K_TOP + k];
        const float* w = wdT + (size_t)c * DM + d0;
        float4 w0 = *(const float4*)(w);
        float4 w1 = *(const float4*)(w + 4);
        acc0.x = fmaf(v, w0.x, acc0.x); acc0.y = fmaf(v, w0.y, acc0.y);
        acc0.z = fmaf(v, w0.z, acc0.z); acc0.w = fmaf(v, w0.w, acc0.w);
        acc1.x = fmaf(v, w1.x, acc1.x); acc1.y = fmaf(v, w1.y, acc1.y);
        acc1.z = fmaf(v, w1.z, acc1.z); acc1.w = fmaf(v, w1.w, acc1.w);
    }
    float* o = x_hat + (size_t)n * DM + d0;
    *(float4*)(o)     = acc0;
    *(float4*)(o + 4) = acc1;
}

// ---------------------------------------------------------------- launcher
extern "C" void kernel_launch(void* const* d_in, const int* in_sizes, int n_in,
                              void* d_out, int out_size, void* d_ws, size_t ws_size,
                              hipStream_t stream) {
    const float* x     = (const float*)d_in[0];
    const float* b_pre = (const float*)d_in[1];
    const float* W_enc = (const float*)d_in[2];
    const float* b_enc = (const float*)d_in[3];
    const float* W_dec = (const float*)d_in[4];
    const float* b_dec = (const float*)d_in[5];

    float* x_hat    = (float*)d_out;                          // [N][DM]
    float* z_sparse = x_hat + (size_t)N_ROWS * DM;            // [N][DS]
    float* z_pre    = z_sparse + (size_t)N_ROWS * DS;         // [N][DS]

    // workspace layout (~402 MB total)
    char* ws = (char*)d_ws;
    auto align_up = [](size_t v) { return (v + 255) & ~(size_t)255; };
    size_t off = 0;
    uint16_t* a_bf = (uint16_t*)(ws + off); off += align_up(2ull * N_ROWS * DM);
    uint16_t* w_bf = (uint16_t*)(ws + off); off += align_up(2ull * DS * DM);
    float*    tv   = (float*)(ws + off);    off += align_up(4ull * N_ROWS * K_TOP);
    int*      ti   = (int*)(ws + off);      off += align_up(4ull * N_ROWS * K_TOP);
    float*    wdT  = (float*)(ws + off);    off += align_up(4ull * DS * DM);
    (void)ws_size; (void)in_sizes; (void)n_in; (void)out_size;

    // Allow 128KB dynamic LDS for the top-k row cache (320KB/WGP on CDNA5).
    static const size_t topkLds = (size_t)DS * sizeof(float);
    (void)hipFuncSetAttribute(reinterpret_cast<const void*>(topk_kernel),
                              hipFuncAttributeMaxDynamicSharedMemorySize,
                              (int)topkLds);

    // 1) bf16 conversions (fused bias-subtract for x)
    {
        long long totX = (long long)N_ROWS * DM;
        convert_bf16_kernel<<<(unsigned)(totX / 8 / 256), 256, 0, stream>>>(
            x, b_pre, (uint32_t*)a_bf, DM, totX);
        long long totW = (long long)DS * DM;
        convert_bf16_kernel<<<(unsigned)(totW / 8 / 256), 256, 0, stream>>>(
            W_enc, nullptr, (uint32_t*)w_bf, DM, totW);
    }

    // 2) W_dec transpose
    {
        dim3 grid(DS / 32, DM / 32), block(32, 8);
        transpose_kernel<<<grid, block, 0, stream>>>(W_dec, wdT);
    }

    // 3) encode GEMM: grid = (S tiles of 128, M tiles of 64)
    {
        dim3 grid(DS / 128, N_ROWS / 64);
        encode_wmma_kernel<<<grid, 256, 0, stream>>>(a_bf, w_bf, b_enc, z_pre);
    }

    // 4) top-K per row (row staged in LDS)
    topk_kernel<<<N_ROWS, 256, topkLds, stream>>>(z_pre, z_sparse, tv, ti);

    // 5) sparse decode
    decode_kernel<<<N_ROWS, 256, 0, stream>>>(wdT, tv, ti, b_dec, x_hat);
}
